// MultiHeadAttention_35433480192372
// MI455X (gfx1250) — matrix operand order repaired and checked by reading, not compiled, not run on hardware
//
#include <hip/hip_runtime.h>

// CDNA5 / gfx1250: wave32, WMMA 16x16x32 bf16 -> f32, TDM staging to LDS.
typedef __attribute__((ext_vector_type(16))) __bf16       v16bf;
typedef __attribute__((ext_vector_type(8)))  __bf16       v8bf;
typedef __attribute__((ext_vector_type(8)))  float        v8f;
typedef __attribute__((ext_vector_type(4)))  unsigned int u32x4;
typedef __attribute__((ext_vector_type(4)))  int          i32x4;
typedef __attribute__((ext_vector_type(8)))  int          i32x8;

#ifndef __has_builtin
#define __has_builtin(x) 0
#endif
#if defined(__gfx1250__)
#if __has_builtin(__builtin_amdgcn_tensor_load_to_lds)
#define USE_TDM 1
#endif
#endif
#ifndef USE_TDM
#define USE_TDM 0
#endif

namespace {
constexpr int kE    = 1024;         // embed
constexpr int kH    = 16;           // heads
constexpr int kD    = 64;           // head dim
constexpr int kN    = 4;            // batch
constexpr int kL    = 4096;         // seq len
constexpr int kRows = kN * kL;      // 16384 rows of x / S
constexpr int kRep  = kL / 256;     // 16x output replication
constexpr int kKPh  = 512;          // K per LDS staging phase (64 KiB of Wv_sum)
}

// ---------------------------------------------------------------------------
// Kernel 1: fold the head-sum into Wv (Wv_sum[d][e] = sum_h Wv[h*64+d][e]),
// convert Wv_sum and Wo to bf16 for the WMMA path.
// ---------------------------------------------------------------------------
__global__ void __launch_bounds__(256)
prep_weights(const float* __restrict__ Wv, const float* __restrict__ Wo,
             __bf16* __restrict__ WvSum, __bf16* __restrict__ WoB) {
  int idx = blockIdx.x * blockDim.x + threadIdx.x;
  if (idx < kD * kE) {
    int d = idx >> 10;
    int e = idx & (kE - 1);
    float s = 0.f;
#pragma unroll
    for (int h = 0; h < kH; ++h) s += Wv[(size_t)(h * kD + d) * kE + e];
    WvSum[idx] = (__bf16)s;              // row-major (64 x 1024)
  }
  int j = idx - kD * kE;
  if (j >= 0 && j < kE * kE) WoB[j] = (__bf16)Wo[j];
}

// ---------------------------------------------------------------------------
// Stage one K-phase of Wv_sum (64 rows x 512 bf16, row stride 1024 bf16) into
// LDS via the Tensor Data Mover. LDS layout: rows packed at stride 512 bf16.
// Issued by wave 0 only (TDM ignores EXEC; one descriptor per phase).
// ---------------------------------------------------------------------------
__device__ __forceinline__ void stage_wvsum_phase(const __bf16* WvSum,
                                                  __bf16* sB, int ph) {
#if USE_TDM
  if (threadIdx.x < 32) {
    unsigned long long ga =
        (unsigned long long)WvSum + (unsigned long long)ph * (kKPh * 2);
    u32x4 g0;
    g0[0] = 1u;                                   // count=1, user D#
    g0[1] = 0u;                                   // lds_addr = 0 (sB is LDS base)
    g0[2] = (unsigned int)ga;                     // global_addr[31:0]
    g0[3] = ((unsigned int)(ga >> 32) & 0x1FFFFFFu) | 0x80000000u; // [56:32]|type=2
    i32x8 g1;
    g1[0] = 0x00020000;        // data_size=2 (4-byte units); no multicast/pad
    g1[1] = (int)(512u << 16); // tensor_dim0 = 512 units (2048 B row)
    g1[2] = (int)(64u << 16);  // tensor_dim0 hi=0 | tensor_dim1 = 64 rows
    g1[3] = (int)(256u << 16); // tensor_dim1 hi=0 | tile_dim0 = 256 units (1024 B)
    g1[4] = 64;                // tile_dim1 = 64 rows, tile_dim2 = 0
    g1[5] = 512;               // tensor_dim0_stride = 512 units
    g1[6] = 0;                 // stride hi | tensor_dim1_stride lo
    g1[7] = 0;
    i32x4 z4 = {0, 0, 0, 0};
#if __clang_major__ >= 23
    i32x8 z8 = {0, 0, 0, 0, 0, 0, 0, 0};
    __builtin_amdgcn_tensor_load_to_lds(g0, g1, z4, z4, z8, 0);
#else
    __builtin_amdgcn_tensor_load_to_lds(g0, g1, z4, z4, 0);
#endif
    __builtin_amdgcn_s_wait_tensorcnt(0);
  }
#else
  // Fallback: cooperative copy (also what the host pass type-checks).
  const unsigned int* src = (const unsigned int*)WvSum;
  unsigned int* dst = (unsigned int*)sB;
  for (int u = threadIdx.x; u < kD * (kKPh / 2); u += 256) {
    int r = u >> 8;          // row (0..63)
    int c = u & 255;         // 4-byte unit within phase row
    dst[u] = src[r * (kE / 2) + ph * (kKPh / 2) + c];
  }
#endif
}

// ---------------------------------------------------------------------------
// Kernel 2: S(16384 x 64) = x(16384 x 1024) @ Wv_sum^T(1024 x 64), bf16 WMMA.
// One wave per 16-row M-tile; 4 N-tiles (D=64) per wave so x streams exactly
// once from HBM. B operand staged to LDS by TDM in two 64 KiB K-phases.
// A layout (16-bit 16x32): lane half 0 -> K 0..15, half 1 -> K 16..31, row = lane&15.
// B layout (32x16): lane half 0 -> K 0..15, half 1 -> K 16..31, column = lane&15.
// ---------------------------------------------------------------------------
__global__ void __launch_bounds__(256)
gemm1_xv(const float* __restrict__ x, const __bf16* __restrict__ WvSumG,
         __bf16* __restrict__ S) {
  __shared__ __bf16 sB[kD * kKPh];   // 64 KiB staging buffer (one K-phase)

  const int lane = threadIdx.x & 31;
  const int wave = threadIdx.x >> 5;
  const int half = lane >> 4;
  const int l15  = lane & 15;
  const int mbase = (blockIdx.x * 8 + wave) * 16;

  const float* xr = x + (size_t)(mbase + l15) * kE;

  v8f acc0 = {}, acc1 = {}, acc2 = {}, acc3 = {};

  for (int ph = 0; ph < 2; ++ph) {
    __syncthreads();                       // previous phase's LDS reads done
    stage_wvsum_phase(WvSumG, sB, ph);
    // The TDM builtin only sees the LDS address as descriptor *data*, so the
    // compiler would otherwise treat sB as never-written and fold the LDS
    // loads below to poison (observed: ds=0, constant B operand). Escape the
    // pointer + declare memory clobbered so the DMA write is modeled.
    asm volatile("" : : "r"((const void*)sB) : "memory");
    __syncthreads();                       // staged data visible to all waves

    const float* xph = xr + ph * kKPh;
    for (int k0 = 0; k0 < kKPh; k0 += 32) {
      // ---- A operand: row (mbase + l15) of x, f32 -> bf16
      const float* xp = xph + k0 + half * 16;
      __builtin_prefetch(xp + 32, 0, 0);   // next k-step of the streamed row
      float4 f0 = *(const float4*)(xp);        // K +0..3
      float4 f1 = *(const float4*)(xp + 4);    // K +4..7
      float4 f2 = *(const float4*)(xp + 8);    // K +8..11 (contiguous layout)
      float4 f3 = *(const float4*)(xp + 12);   // K +12..15
      v16bf a;
      a[0] = (__bf16)f0.x;  a[1] = (__bf16)f0.y;  a[2] = (__bf16)f0.z;  a[3] = (__bf16)f0.w;
      a[4] = (__bf16)f1.x;  a[5] = (__bf16)f1.y;  a[6] = (__bf16)f1.z;  a[7] = (__bf16)f1.w;
      a[8] = (__bf16)f2.x;  a[9] = (__bf16)f2.y;  a[10] = (__bf16)f2.z; a[11] = (__bf16)f2.w;
      a[12] = (__bf16)f3.x; a[13] = (__bf16)f3.y; a[14] = (__bf16)f3.z; a[15] = (__bf16)f3.w;

      // ---- B operands from LDS: lane reads row (tile*16 + l15), stride 512
      const __bf16* bb = sB + (size_t)l15 * kKPh + k0 + half * 16;
      v16bf b0 = *(const v16bf*)(bb);
      v16bf b1 = *(const v16bf*)(bb + (size_t)16 * kKPh);
      v16bf b2 = *(const v16bf*)(bb + (size_t)32 * kKPh);
      v16bf b3 = *(const v16bf*)(bb + (size_t)48 * kKPh);

      acc0 = __builtin_amdgcn_wmma_f32_16x16x32_bf16(false, a, false, b0, (short)0, acc0, false, false);
      acc1 = __builtin_amdgcn_wmma_f32_16x16x32_bf16(false, a, false, b1, (short)0, acc1, false, false);
      acc2 = __builtin_amdgcn_wmma_f32_16x16x32_bf16(false, a, false, b2, (short)0, acc2, false, false);
      acc3 = __builtin_amdgcn_wmma_f32_16x16x32_bf16(false, a, false, b3, (short)0, acc3, false, false);
    }
  }

  // C/D layout: VGPR r -> M = r + 8*half, N = l15. Store S as bf16 (feeds GEMM2).
#pragma unroll
  for (int r = 0; r < 8; ++r) {
    int row = mbase + r + half * 8;
    __bf16* sp = S + (size_t)row * kD + l15;
    sp[0]  = (__bf16)acc0[r];
    sp[16] = (__bf16)acc1[r];
    sp[32] = (__bf16)acc2[r];
    sp[48] = (__bf16)acc3[r];
  }
}

// ---------------------------------------------------------------------------
// Kernel 3: T(1024 x 1024) = U @ Wo^T + bo, where U is the S buffer
// reinterpreted as (1024 x 1024) row-major (the reference's reshape quirk makes
// this exact). Each wave computes one 16x16 tile and writes it to all 16
// replicated output positions (out[n, l2, f] = T[n*256 + (l2 & 255), f]).
// Both operands (2 MiB each) are L2-resident on a 192 MB L2.
// ---------------------------------------------------------------------------
__global__ void __launch_bounds__(256)
gemm2_out(const __bf16* __restrict__ U, const __bf16* __restrict__ WoB,
          const float* __restrict__ bo, float* __restrict__ out) {
  const int lane = threadIdx.x & 31;
  const int wave = threadIdx.x >> 5;
  const int half = lane >> 4;
  const int l15  = lane & 15;
  const int g    = blockIdx.x * 8 + wave;   // 0..4095 tiles (64 x 64)
  const int rbase = (g >> 6) * 16;          // M-tile base row in U
  const int f     = (g & 63) * 16 + l15;    // output column

  const __bf16* ur = U + (size_t)(rbase + l15) * kE;
  const __bf16* wr = WoB + (size_t)f * kE;

  v8f acc = {};
  for (int k0 = 0; k0 < kE; k0 += 32) {
    const __bf16* up = ur + k0 + half * 16;
    v8bf a0 = *(const v8bf*)(up);
    v8bf a1 = *(const v8bf*)(up + 8);
    v16bf a = __builtin_shufflevector(a0, a1, 0, 1, 2, 3, 4, 5, 6, 7,
                                      8, 9, 10, 11, 12, 13, 14, 15);
    v16bf b = *(const v16bf*)(wr + k0 + half * 16);
    acc = __builtin_amdgcn_wmma_f32_16x16x32_bf16(false, a, false, b, (short)0, acc, false, false);
  }

  const float bof = bo[f];
#pragma unroll
  for (int r = 0; r < 8; ++r) {
    int rowU = rbase + r + half * 8;        // n*256 + l2'
    int n   = rowU >> 8;
    int l2p = rowU & 255;
    float val = acc[r] + bof;
    float* op = out + ((size_t)n * kL + l2p) * kE + f;
#pragma unroll
    for (int j = 0; j < kRep; ++j)          // 16x replication along L
      __builtin_nontemporal_store(val, op + (size_t)j * 256 * kE);
  }
}

// ---------------------------------------------------------------------------
extern "C" void kernel_launch(void* const* d_in, const int* in_sizes, int n_in,
                              void* d_out, int out_size, void* d_ws, size_t ws_size,
                              hipStream_t stream) {
  (void)in_sizes; (void)n_in; (void)out_size; (void)ws_size;
  const float* x  = (const float*)d_in[0];
  // d_in[1] = Wq, d_in[2] = Wk: provably unused (softmax rows sum to 1 and the
  // degenerate einsum makes the output independent of attention).
  const float* Wv = (const float*)d_in[3];
  const float* Wo = (const float*)d_in[4];
  const float* bo = (const float*)d_in[5];
  float* out = (float*)d_out;

  char* ws = (char*)d_ws;
  __bf16* WvSum = (__bf16*)ws;                                   // 128 KiB
  __bf16* WoB   = (__bf16*)(ws + 128 * 1024);                    // 2 MiB
  __bf16* S     = (__bf16*)(ws + 128 * 1024 + 2 * 1024 * 1024);  // 2 MiB

  const int prepElems = kD * kE + kE * kE;
  prep_weights<<<(prepElems + 255) / 256, 256, 0, stream>>>(Wv, Wo, WvSum, WoB);

  // 16384 rows / (8 waves * 16 rows) = 128 blocks
  gemm1_xv<<<kRows / 128, 256, 0, stream>>>(x, WvSum, S);

  // 4096 tiles / 8 waves = 512 blocks
  gemm2_out<<<512, 256, 0, stream>>>(S, WoB, bo, out);
}
